// FixedGraphAttentionLayer_82523501625840
// MI455X (gfx1250) — matrix operand order repaired and checked by reading, not compiled, not run on hardware
//
#include <hip/hip_runtime.h>
#include <hip/hip_bf16.h>
#include <stdint.h>

typedef __attribute__((ext_vector_type(16))) _Float16 v16h;
typedef __attribute__((ext_vector_type(8)))  _Float16 v8h;
typedef __attribute__((ext_vector_type(8)))  float    v8f;
typedef __attribute__((ext_vector_type(4)))  float    v4f;
typedef __attribute__((ext_vector_type(4)))  int      v4i;

#define B_      8
#define N_      2048
#define D_      128
#define NEGINF_ (-1.0e9f)
#define JT      64          // j tile width in fused attention
#define ESTRIDE 68          // LDS row stride (floats); 16B-aligned rows for v4f
#define HSTRIDE 72          // LDS row stride (halfs) for hT tile: 144B, 16B-aligned
#define WAVES   8           // waves per block (wave32)

// K index inside a 16x32 f16 A-fragment for half-slot h (0..15), lane group g (0/1).
// CDNA5 ISA 16-bit A-matrix layout: V0..V3 hold K=0..7 (lanes0-15) / K=8..15
// (lanes16-31); V4..V7 hold K=16..23 / 24..31.
__device__ __forceinline__ int a_k_index(int h, int g) {
    int v = h >> 1, pos = h & 1;
    int base = (v < 4) ? 0 : 16;
    return base + (v & 3) * 2 + pos + g * 8;
}

// ---------------------------------------------------------------------------
// Kernel 1: h = x @ W^T  (f16 WMMA, f32 accumulate), store TRANSPOSED f16
// hT[b][d][n] so the attention GEMM B-fragments are contiguous loads.
// One wave computes 16 rows x 128 cols.
// ---------------------------------------------------------------------------
__global__ __launch_bounds__(32 * WAVES) void k_gemm_h(
        const float* __restrict__ x, const float* __restrict__ W,
        _Float16* __restrict__ hT) {
    int lane = threadIdx.x & 31;
    int wave = threadIdx.x >> 5;
    int tile = blockIdx.x * WAVES + wave;        // 0 .. (B*N/16 - 1)
    int row0 = tile * 16;
    int b  = row0 / N_;
    int n0 = row0 % N_;
    int m  = lane & 15;
    int g  = lane >> 4;

    v16h afr[4];
    const float* xrow = x + (size_t)(row0 + m) * D_;
    #pragma unroll
    for (int ks = 0; ks < 4; ++ks) {
        #pragma unroll
        for (int h = 0; h < 16; ++h)
            afr[ks][h] = (_Float16)xrow[ks * 32 + g * 16 + h];
    }

    #pragma unroll
    for (int t = 0; t < 8; ++t) {
        int d = t * 16 + m;
        const float* wrow = W + (size_t)d * D_;   // B[k][d] = W[d][k]
        v8f acc = {};
        #pragma unroll
        for (int ks = 0; ks < 4; ++ks) {
            v16h bfr;                             // lane holds K = g*16 .. g*16+15
            int kb = ks * 32 + g * 16;
            #pragma unroll
            for (int h = 0; h < 16; ++h)
                bfr[h] = (_Float16)wrow[kb + h];
            acc = __builtin_amdgcn_wmma_f32_16x16x32_f16(
                    false, afr[ks], false, bfr, (short)0, acc, false, false);
        }
        v8h o;                                    // C: lane holds rows m = g*8 + r
        #pragma unroll
        for (int r = 0; r < 8; ++r) o[r] = (_Float16)acc[r];
        *(v8h*)(hT + ((size_t)(b * D_ + d) * N_ + n0 + g * 8)) = o;
    }
}

// ---------------------------------------------------------------------------
// Kernel 2: s_i = h.a1, s_j = h.a2 from transposed hT (coalesced over n).
// ---------------------------------------------------------------------------
__global__ void k_scores(const _Float16* __restrict__ hT,
                         const float* __restrict__ a1, const float* __restrict__ a2,
                         float* __restrict__ s1, float* __restrict__ s2) {
    int idx = blockIdx.x * blockDim.x + threadIdx.x;
    if (idx >= B_ * N_) return;
    int b = idx / N_, n = idx % N_;
    const _Float16* base = hT + (size_t)b * D_ * N_ + n;
    float r1 = 0.f, r2 = 0.f;
    #pragma unroll 4
    for (int d = 0; d < D_; ++d) {
        float hv = (float)base[(size_t)d * N_];
        r1 += hv * a1[d];
        r2 += hv * a2[d];
    }
    s1[idx] = r1; s2[idx] = r2;
}

// ---------------------------------------------------------------------------
// Kernel 3: fused masked softmax (online) + alpha @ h (WMMA) + residual + LN.
// One wave owns a 16-row i-tile; adj read exactly once from HBM with b128
// loads. The hT j-tile (shared by all 8 waves of the block) is staged into
// LDS with GLOBAL_LOAD_ASYNC_TO_LDS_B128, overlapped with adj/softmax phases.
// ---------------------------------------------------------------------------
__global__ __launch_bounds__(32 * WAVES) void k_attn(
        const float* __restrict__ x, const int* __restrict__ adj,
        const _Float16* __restrict__ hT,
        const float* __restrict__ s1, const float* __restrict__ s2,
        const float* __restrict__ gamma, const float* __restrict__ beta,
        float* __restrict__ out) {
    __shared__ float    smem[WAVES * (16 * ESTRIDE + 16)];   // 35328 B
    __shared__ _Float16 hTile[D_ * HSTRIDE];                 // 18432 B

    int lane = threadIdx.x & 31;
    int wave = threadIdx.x >> 5;
    float* E    = smem + wave * (16 * ESTRIDE + 16);  // 16 x JT scores/probs
    float* rbuf = E + 16 * ESTRIDE;                   // per-row broadcast slots

    int tile = blockIdx.x * WAVES + wave;             // 0 .. (B*N/16 - 1)
    int b  = (blockIdx.x * WAVES) / (N_ / 16);        // all waves: same batch
    int i0 = (tile % (N_ / 16)) * 16;
    int m16 = lane & 15;
    int g   = lane >> 4;

    v8f acc[8];
    #pragma unroll
    for (int t = 0; t < 8; ++t) acc[t] = (v8f){};

    float m_run = -3.0e38f, l_run = 0.f;

    const int*      adj_b = adj + (size_t)b * N_ * N_;
    const float*    s1_b  = s1 + b * N_;
    const float*    s2_b  = s2 + b * N_;
    const _Float16* hT_b  = hT + (size_t)b * D_ * N_;
    const uint32_t  hbase = (uint32_t)(uintptr_t)&hTile[0];

    for (int j0 = 0; j0 < N_; j0 += JT) {
        __syncthreads();   // previous tile's LDS reads complete everywhere

        // ---- async DMA: stage hT[:, j0:j0+64] into LDS (1024 B128 chunks,
        //      4 per thread), ASYNCcnt-tracked; overlaps phases below ----
        #pragma unroll
        for (int k = 0; k < 4; ++k) {
            int c    = (int)threadIdx.x + k * 256;    // 0..1023
            int d    = c >> 3;
            int part = c & 7;
            const _Float16* gsrc = hT_b + (size_t)d * N_ + j0 + part * 8;
            uint32_t laddr = hbase + (uint32_t)(d * HSTRIDE + part * 8) * 2u;
            asm volatile("global_load_async_to_lds_b128 %0, %1, off"
                         :: "v"(laddr), "v"((unsigned long long)(uintptr_t)gsrc)
                         : "memory");
        }

        // ---- masked LeakyReLU scores into LDS; adj streamed with b128 ----
        #pragma unroll
        for (int t = 0; t < 8; ++t) {
            int idx4 = t * 32 + lane;                 // 256 float4 slots
            int mm  = idx4 >> 4;                      // row 0..15
            int jj4 = (idx4 & 15) * 4;                // col 0,4,..,60
            v4i a   = *(const v4i*)(adj_b + (size_t)(i0 + mm) * N_ + j0 + jj4);
            v4f s2v = *(const v4f*)(s2_b + j0 + jj4);
            float s1v = s1_b[i0 + mm];
            v4f e;
            #pragma unroll
            for (int c = 0; c < 4; ++c) {
                float v = s1v + s2v[c];
                v = (v >= 0.f) ? v : 0.2f * v;        // LeakyReLU(0.2)
                e[c] = (a[c] == 0) ? NEGINF_ : v;
            }
            *(v4f*)(E + mm * ESTRIDE + jj4) = e;
        }
        if (j0 + JT < N_)
            __builtin_prefetch(&adj_b[(size_t)(i0 + m16) * N_ + j0 + JT + g * 32], 0, 1);
        __syncthreads();

        // ---- online softmax: lanes (l, l+16) co-own row l%16; the 32 scores
        //      are pulled into registers once (v4f) and reused for exp ----
        {
            int row = m16;
            int jo  = g * 32;
            v4f ev[8];
            const v4f* Er = (const v4f*)(E + row * ESTRIDE + jo);
            #pragma unroll
            for (int k = 0; k < 8; ++k) ev[k] = Er[k];
            float lmax = -3.0e38f;
            #pragma unroll
            for (int k = 0; k < 8; ++k)
                #pragma unroll
                for (int c = 0; c < 4; ++c)
                    lmax = fmaxf(lmax, ev[k][c]);
            lmax = fmaxf(lmax, __shfl_xor(lmax, 16));
            float m_new = fmaxf(m_run, lmax);
            float scale = __expf(m_run - m_new);
            float psum = 0.f;
            v4f* Ew = (v4f*)(E + row * ESTRIDE + jo);
            #pragma unroll
            for (int k = 0; k < 8; ++k) {
                v4f p;
                #pragma unroll
                for (int c = 0; c < 4; ++c) {
                    p[c] = __expf(ev[k][c] - m_new);
                    psum += p[c];
                }
                Ew[k] = p;
            }
            psum += __shfl_xor(psum, 16);
            l_run = l_run * scale + psum;
            m_run = m_new;
            if (lane < 16) rbuf[row] = scale;
        }

        // hT tile must be fully in LDS (all waves) before the WMMA phase
        asm volatile("s_wait_asynccnt 0x0" ::: "memory");
        __syncthreads();

        // ---- rescale accumulators by exp(m_old - m_new) per row ----
        float rr[8];
        #pragma unroll
        for (int r = 0; r < 8; ++r) rr[r] = rbuf[g * 8 + r];
        #pragma unroll
        for (int t = 0; t < 8; ++t)
            #pragma unroll
            for (int r = 0; r < 8; ++r) acc[t][r] *= rr[r];

        // ---- P(16x64) @ H(64x128): 2 k-steps x 8 d-tiles of WMMA,
        //      B-fragments from the LDS-staged hT tile ----
        #pragma unroll
        for (int ks = 0; ks < 2; ++ks) {
            v16h afr;
            #pragma unroll
            for (int h = 0; h < 16; ++h)
                afr[h] = (_Float16)E[m16 * ESTRIDE + ks * 32 + g * 16 + h];
            #pragma unroll
            for (int t = 0; t < 8; ++t) {
                int d = t * 16 + m16;
                int off = d * HSTRIDE + ks * 32 + g * 16;   // 16B-aligned
                v8h lo = *(const v8h*)(hTile + off);
                v8h hi = *(const v8h*)(hTile + off + 8);
                v16h bfr = __builtin_shufflevector(lo, hi,
                        0,1,2,3,4,5,6,7,8,9,10,11,12,13,14,15);
                acc[t] = __builtin_amdgcn_wmma_f32_16x16x32_f16(
                        false, afr, false, bfr, (short)0, acc[t], false, false);
            }
        }
    }

    // ---- broadcast softmax denominators ----
    __syncthreads();
    if (lane < 16) rbuf[m16] = l_run;
    __syncthreads();
    float invl[8];
    #pragma unroll
    for (int r = 0; r < 8; ++r) invl[r] = 1.0f / rbuf[g * 8 + r];

    // ---- finalize: divide, residual, LayerNorm (16-lane shuffle reduce) ----
    float ssum[8], ssq[8];
    #pragma unroll
    for (int r = 0; r < 8; ++r) { ssum[r] = 0.f; ssq[r] = 0.f; }
    #pragma unroll
    for (int r = 0; r < 8; ++r) {
        int i = i0 + g * 8 + r;
        const float* xrow = x + (size_t)(b * N_ + i) * D_;
        #pragma unroll
        for (int t = 0; t < 8; ++t) {
            int d = t * 16 + m16;
            float v = acc[t][r] * invl[r] + xrow[d];
            acc[t][r] = v;
            ssum[r] += v;
            ssq[r]  += v * v;
        }
    }
    #pragma unroll
    for (int r = 0; r < 8; ++r) {
        #pragma unroll
        for (int off = 1; off < 16; off <<= 1) {
            ssum[r] += __shfl_xor(ssum[r], off);
            ssq[r]  += __shfl_xor(ssq[r], off);
        }
    }
    #pragma unroll
    for (int r = 0; r < 8; ++r) {
        int i = i0 + g * 8 + r;
        float mu  = ssum[r] * (1.0f / D_);
        float var = ssq[r] * (1.0f / D_) - mu * mu;
        float rs  = rsqrtf(var + 1e-5f);
        float* orow = out + (size_t)(b * N_ + i) * D_;
        #pragma unroll
        for (int t = 0; t < 8; ++t) {
            int d = t * 16 + m16;
            orow[d] = (acc[t][r] - mu) * rs * gamma[d] + beta[d];
        }
    }
}

extern "C" void kernel_launch(void* const* d_in, const int* in_sizes, int n_in,
                              void* d_out, int out_size, void* d_ws, size_t ws_size,
                              hipStream_t stream) {
    (void)in_sizes; (void)n_in; (void)out_size; (void)ws_size;
    const float* x     = (const float*)d_in[0];
    const int*   adj   = (const int*)  d_in[1];
    const float* W     = (const float*)d_in[2];
    const float* a1    = (const float*)d_in[3];
    const float* a2    = (const float*)d_in[4];
    const float* gamma = (const float*)d_in[5];
    const float* beta  = (const float*)d_in[6];
    float* out = (float*)d_out;

    // workspace: hT (f16, transposed h) | s1 | s2
    _Float16* hT = (_Float16*)d_ws;
    float* s1 = (float*)((char*)d_ws + (size_t)B_ * D_ * N_ * sizeof(_Float16));
    float* s2 = s1 + B_ * N_;

    int tiles = (B_ * N_) / 16;                  // 1024 i-tiles
    k_gemm_h<<<tiles / WAVES, 32 * WAVES, 0, stream>>>(x, W, hT);
    k_scores<<<(B_ * N_ + 255) / 256, 256, 0, stream>>>(hT, a1, a2, s1, s2);
    k_attn<<<tiles / WAVES, 32 * WAVES, 0, stream>>>(x, adj, hT, s1, s2,
                                                     gamma, beta, out);
}
